// _ConvolutionModule_43267500540319
// MI455X (gfx1250) — matrix operand order repaired and checked by reading, not compiled, not run on hardware
//
#include <hip/hip_runtime.h>
#include <hip/hip_bf16.h>

// ---- CDNA5 WMMA types -------------------------------------------------------
typedef __attribute__((ext_vector_type(16))) __bf16 v16bf;
typedef __attribute__((ext_vector_type(8)))  float  v8f;
typedef __attribute__((ext_vector_type(4)))  int    v4i;

union FragBF { v16bf v; unsigned int u[8]; };
union FragF  { v8f   v; float        f[8]; };

__device__ inline unsigned short f2bf(float f) {
  unsigned int u = __builtin_bit_cast(unsigned int, f);
  u += 0x7FFFu + ((u >> 16) & 1u);          // round-to-nearest-even
  return (unsigned short)(u >> 16);
}

// ---- CDNA5 async global->LDS copy (ASYNCcnt path) ---------------------------
__device__ inline void async_cp16(const unsigned int* g, unsigned int* l) {
#if __has_builtin(__builtin_amdgcn_global_load_async_to_lds_b128)
  __builtin_amdgcn_global_load_async_to_lds_b128((v4i*)g, (v4i*)l, 0, 0);
#else
  unsigned int loff = (unsigned int)(unsigned long long)
      (__attribute__((address_space(3))) unsigned int*)l;
  asm volatile("global_load_async_to_lds_b128 %0, %1, off"
               :: "v"(loff), "v"((unsigned long long)g) : "memory");
#endif
}
__device__ inline void wait_async0() {
#if __has_builtin(__builtin_amdgcn_s_wait_asynccnt)
  __builtin_amdgcn_s_wait_asynccnt(0);
#else
  asm volatile("s_wait_asynccnt 0x0" ::: "memory");
#endif
}

// Problem constants
#define DD        512
#define M_TOTAL   81920      // 2560*32 rows
#define RC_ROWS   16384      // 512*32 rows (right-context part)
#define OUT_RC    33554432   // 2048*32*512  (utterance output size)
#define OUT_NS    41943040   // + 512*32*512 (rc output size)
#define XWS_OFF   83886080u  // bf16 activation buffer: 81920*512*2
#define W1BF_OFF  251658240u // 83886080 + 167772160
#define W2BF_OFF  252706816u // + 1024*512*2

// ---- K-: fp32 -> bf16 weight pre-conversion (runs once per call, tiny) ------
__global__ __launch_bounds__(256) void cvt_bf16_kernel(
    const float* __restrict__ src, unsigned short* __restrict__ dst, int n) {
  int i = blockIdx.x * 256 + threadIdx.x;
  if (i < n) dst[i] = f2bf(src[i]);
}

// ---- K0: LayerNorm1 -> bf16 activations ------------------------------------
__global__ __launch_bounds__(256) void ln1_kernel(
    const float* __restrict__ utt, const float* __restrict__ rc,
    const float* __restrict__ g,   const float* __restrict__ bb,
    unsigned short* __restrict__ a1) {
  int r = blockIdx.x;
  const float* row = (r < RC_ROWS) ? (rc + (size_t)r * DD)
                                   : (utt + (size_t)(r - RC_ROWS) * DD);
  int tid = threadIdx.x;
  float x0 = row[tid], x1 = row[tid + 256];
  __shared__ float s1[256], s2[256];
  s1[tid] = x0 + x1;
  s2[tid] = x0 * x0 + x1 * x1;
  __syncthreads();
  for (int st = 128; st > 0; st >>= 1) {
    if (tid < st) { s1[tid] += s1[tid + st]; s2[tid] += s2[tid + st]; }
    __syncthreads();
  }
  float m   = s1[0] * (1.0f / 512.0f);
  float var = s2[0] * (1.0f / 512.0f) - m * m;
  float rs  = rsqrtf(var + 1e-5f);
  a1[(size_t)r * DD + tid]       = f2bf((x0 - m) * rs * g[tid]       + bb[tid]);
  a1[(size_t)r * DD + tid + 256] = f2bf((x1 - m) * rs * g[tid + 256] + bb[tid + 256]);
}

// ---- K1: GEMM1 (81920x512 . 512x1024) + bias + GLU -> fp32 x ---------------
// Block tile 64(M) x 64(N); per wave one 16x32 A-frag, 4 accumulators
// (a-half/g-half x 2 n-subtiles) so GLU fuses in registers.
// Double-buffered LDS filled by async global->LDS b128 copies.
__global__ __launch_bounds__(256) void gemm1_kernel(
    const unsigned short* __restrict__ a1, const unsigned short* __restrict__ w1bf,
    const float* __restrict__ b1, float* __restrict__ xws) {
  __shared__ __attribute__((aligned(16))) unsigned int At[2][64 * 20];
  __shared__ __attribute__((aligned(16))) unsigned int Bt[2][128 * 20];
  int tid = threadIdx.x;
  int m0 = blockIdx.x * 64;
  int n0 = blockIdx.y * 64;
  int lane = tid & 31;
  int w    = tid >> 5;
  int wm = w & 3, wn = w >> 2;
  int lm = lane & 15;
  int lg = lane >> 4;
  const unsigned int* a1u = (const unsigned int*)a1;
  const unsigned int* w1u = (const unsigned int*)w1bf;

  // per-thread staging chunk (16B): A tile 64x32bf16 = 256 chunks
  int srow = tid >> 2, sc4 = (tid & 3) * 4;

  auto stage = [&](int it, int buf) {
    int ku = it * 16;   // uint offset of K-slice
    async_cp16(a1u + (size_t)(m0 + srow) * 256 + ku + sc4,
               &At[buf][srow * 20 + sc4]);
    async_cp16(w1u + (size_t)(n0 + srow) * 256 + ku + sc4,
               &Bt[buf][srow * 20 + sc4]);
    async_cp16(w1u + (size_t)(512 + n0 + srow) * 256 + ku + sc4,
               &Bt[buf][(64 + srow) * 20 + sc4]);
  };

  v8f zero = {0.f,0.f,0.f,0.f,0.f,0.f,0.f,0.f};
  v8f acc[4];
  for (int i = 0; i < 4; i++) acc[i] = zero;

  stage(0, 0);
#pragma unroll 2
  for (int it = 0; it < 16; ++it) {
    int buf = it & 1;
    wait_async0();
    __syncthreads();
    if (it < 15) stage(it + 1, buf ^ 1);

    FragBF af;
    int ar = 16 * wm + lm;
    for (int i = 0; i < 8; i++)
      af.u[i] = At[buf][ar * 20 + 4 * lg + ((i < 4) ? i : i + 4)];
    for (int ns = 0; ns < 4; ns++) {
      int half = ns >> 1, nsub = ns & 1;
      int br = 64 * half + 32 * wn + 16 * nsub + lm;
      FragBF bf;
      for (int i = 0; i < 8; i++)
        bf.u[i] = Bt[buf][br * 20 + 4 * lg + ((i < 4) ? i : i + 4)];
      acc[ns] = __builtin_amdgcn_wmma_f32_16x16x32_bf16(
          false, af.v, false, bf.v, (short)0, acc[ns], false, false);
    }
  }

  for (int nsub = 0; nsub < 2; nsub++) {
    int n = n0 + 32 * wn + 16 * nsub + lm;
    float ba = b1[n], bg = b1[n + 512];
    FragF fa, fg;
    fa.v = acc[nsub];
    fg.v = acc[2 + nsub];
    for (int v = 0; v < 8; v++) {
      int m = m0 + 16 * wm + v + 8 * lg;
      float a  = fa.f[v] + ba;
      float gg = fg.f[v] + bg;
      float sig = 1.0f / (1.0f + __expf(-gg));
      xws[(size_t)m * DD + n] = a * sig;
    }
  }
}

// ---- K2: depthwise conv (causal + segmented rc) + LN2 + SiLU -> bf16 -------
__global__ __launch_bounds__(256) void conv_ln2_kernel(
    const float* __restrict__ xws, const float* __restrict__ state,
    const float* __restrict__ cw,  const float* __restrict__ cb,
    const float* __restrict__ g2,  const float* __restrict__ bb2,
    unsigned short* __restrict__ a2) {
  int r = blockIdx.x;
  int t = r >> 5, b = r & 31;
  int tid = threadIdx.x;
  float y[2];
  for (int c = 0; c < 2; c++) {
    int d = tid + 256 * c;
    float acc = cb[d];
    if (t >= 512) {                 // utterance: state_x = [state(30) | x_utt]
      int tau = t - 512;
      for (int k = 0; k < 31; k++) {
        int j = tau + k;
        float src = (j < 30) ? state[((size_t)b * 512 + d) * 30 + j]
                             : xws[(size_t)(r + 32 * (k - 30)) * DD + d];
        acc += cw[d * 31 + k] * src;
      }
    } else {                        // right-context segment s, pads from x_utt
      int s = t >> 5, p = t & 31;
      for (int k = 0; k < 31; k++) {
        int q = p + k;
        int xrow = (q < 30) ? ((512 + 128 * (s + 1) + q - 30) * 32 + b)
                            : ((s * 32 + q - 30) * 32 + b);
        acc += cw[d * 31 + k] * xws[(size_t)xrow * DD + d];
      }
    }
    y[c] = acc;
  }
  __shared__ float s1[256], s2[256];
  s1[tid] = y[0] + y[1];
  s2[tid] = y[0] * y[0] + y[1] * y[1];
  __syncthreads();
  for (int st = 128; st > 0; st >>= 1) {
    if (tid < st) { s1[tid] += s1[tid + st]; s2[tid] += s2[tid + st]; }
    __syncthreads();
  }
  float m   = s1[0] * (1.0f / 512.0f);
  float var = s2[0] * (1.0f / 512.0f) - m * m;
  float rs  = rsqrtf(var + 1e-5f);
  for (int c = 0; c < 2; c++) {
    int d = tid + 256 * c;
    float z  = (y[c] - m) * rs * g2[d] + bb2[d];
    float sl = z / (1.0f + __expf(-z));   // SiLU
    a2[(size_t)r * DD + d] = f2bf(sl);
  }
}

// ---- K3: GEMM2 (81920x512 . 512x512) + bias + residual -> d_out ------------
__global__ __launch_bounds__(256) void gemm2_kernel(
    const unsigned short* __restrict__ a2, const unsigned short* __restrict__ w2bf,
    const float* __restrict__ b2, const float* __restrict__ utt,
    const float* __restrict__ rc, float* __restrict__ out) {
  __shared__ __attribute__((aligned(16))) unsigned int At[2][64 * 20];
  __shared__ __attribute__((aligned(16))) unsigned int Bt[2][64 * 20];
  int tid = threadIdx.x;
  int m0 = blockIdx.x * 64;
  int n0 = blockIdx.y * 64;
  int lane = tid & 31;
  int w    = tid >> 5;
  int wm = w & 3, wn = w >> 2;
  int lm = lane & 15;
  int lg = lane >> 4;
  const unsigned int* a2u = (const unsigned int*)a2;
  const unsigned int* w2u = (const unsigned int*)w2bf;

  int srow = tid >> 2, sc4 = (tid & 3) * 4;

  auto stage = [&](int it, int buf) {
    int ku = it * 16;
    async_cp16(a2u + (size_t)(m0 + srow) * 256 + ku + sc4,
               &At[buf][srow * 20 + sc4]);
    async_cp16(w2u + (size_t)(n0 + srow) * 256 + ku + sc4,
               &Bt[buf][srow * 20 + sc4]);
  };

  v8f zero = {0.f,0.f,0.f,0.f,0.f,0.f,0.f,0.f};
  v8f acc[2];
  acc[0] = zero; acc[1] = zero;

  stage(0, 0);
#pragma unroll 2
  for (int it = 0; it < 16; ++it) {
    int buf = it & 1;
    wait_async0();
    __syncthreads();
    if (it < 15) stage(it + 1, buf ^ 1);

    FragBF af;
    int ar = 16 * wm + lm;
    for (int i = 0; i < 8; i++)
      af.u[i] = At[buf][ar * 20 + 4 * lg + ((i < 4) ? i : i + 4)];
    for (int ns = 0; ns < 2; ns++) {
      int br = 32 * wn + 16 * ns + lm;
      FragBF bf;
      for (int i = 0; i < 8; i++)
        bf.u[i] = Bt[buf][br * 20 + 4 * lg + ((i < 4) ? i : i + 4)];
      acc[ns] = __builtin_amdgcn_wmma_f32_16x16x32_bf16(
          false, af.v, false, bf.v, (short)0, acc[ns], false, false);
    }
  }

  for (int ns = 0; ns < 2; ns++) {
    int n = n0 + 32 * wn + 16 * ns + lm;
    float bias = b2[n];
    FragF fc; fc.v = acc[ns];
    for (int v = 0; v < 8; v++) {
      int m = m0 + 16 * wm + v + 8 * lg;
      float val = fc.f[v] + bias;
      if (m < RC_ROWS) {             // right-context rows -> second output slab
        val += rc[(size_t)m * DD + n];
        out[(size_t)OUT_RC + (size_t)m * DD + n] = val;
      } else {                       // utterance rows -> first output slab
        int mu = m - RC_ROWS;
        val += utt[(size_t)mu * DD + n];
        out[(size_t)mu * DD + n] = val;
      }
    }
  }
}

// ---- K4: new_state[b,d,j] = x[512+2018+j, b, d] ----------------------------
__global__ __launch_bounds__(256) void state_kernel(
    const float* __restrict__ xws, float* __restrict__ out) {
  int idx = blockIdx.x * 256 + threadIdx.x;   // < 491520 exactly
  int j  = idx % 30;
  int bd = idx / 30;
  int d  = bd & 511;
  int b  = bd >> 9;
  out[(size_t)OUT_NS + idx] = xws[(size_t)((2530 + j) * 32 + b) * DD + d];
}

// ---- host ------------------------------------------------------------------
extern "C" void kernel_launch(void* const* d_in, const int* in_sizes, int n_in,
                              void* d_out, int out_size, void* d_ws, size_t ws_size,
                              hipStream_t stream) {
  const float* utt  = (const float*)d_in[0];
  const float* rc   = (const float*)d_in[1];
  const float* st   = (const float*)d_in[2];
  const float* ln1g = (const float*)d_in[3];
  const float* ln1b = (const float*)d_in[4];
  const float* w1   = (const float*)d_in[5];
  const float* b1   = (const float*)d_in[6];
  const float* cw   = (const float*)d_in[7];
  const float* cb   = (const float*)d_in[8];
  const float* ln2g = (const float*)d_in[9];
  const float* ln2b = (const float*)d_in[10];
  const float* w2   = (const float*)d_in[11];
  const float* b2   = (const float*)d_in[12];
  float* out = (float*)d_out;

  unsigned short* abf  = (unsigned short*)d_ws;                      // bf16 acts (reused)
  float*          xws  = (float*)((char*)d_ws + XWS_OFF);            // fp32 GLU output
  unsigned short* w1bf = (unsigned short*)((char*)d_ws + W1BF_OFF);  // bf16 weights
  unsigned short* w2bf = (unsigned short*)((char*)d_ws + W2BF_OFF);

  cvt_bf16_kernel<<<2048, 256, 0, stream>>>(w1, w1bf, 1024 * 512);
  cvt_bf16_kernel<<<1024, 256, 0, stream>>>(w2, w2bf, 512 * 512);
  ln1_kernel     <<<M_TOTAL, 256, 0, stream>>>(utt, rc, ln1g, ln1b, abf);
  gemm1_kernel   <<<dim3(M_TOTAL / 64, 8), 256, 0, stream>>>(abf, w1bf, b1, xws);
  conv_ln2_kernel<<<M_TOTAL, 256, 0, stream>>>(xws, st, cw, cb, ln2g, ln2b, abf);
  gemm2_kernel   <<<dim3(M_TOTAL / 64, 8), 256, 0, stream>>>(abf, w2bf, b2, utt, rc, out);
  state_kernel   <<<1920, 256, 0, stream>>>(xws, out);
}
